// RecurrentModelWithCRF_20873541059078
// MI455X (gfx1250) — compile-verified, matrix operand order checked by reading, not run on hardware
//
#include <hip/hip_runtime.h>
#include <hip/hip_bf16.h>
#include <math.h>

typedef __bf16 bf16;
typedef __attribute__((ext_vector_type(16))) __bf16 v16bf;
typedef __attribute__((ext_vector_type(8)))  float   v8f;

#define Bsz    32
#define Tsz    2048
#define Dsz    256
#define Hsz    256
#define Gsz    1024      // 4*H
#define Ksz    8
#define HOUTsz 512

// ---------------- helpers ----------------

__device__ __forceinline__ float sigf(float x) { return 1.0f / (1.0f + __expf(-x)); }

// Load a 16x32 bf16 fragment (A layout; B uses the same pattern when the
// operand is stored (N,K) row-major) from a row-major source with stride ld.
// 16-bit layout: lanes 0-15 row r, k in {0..7,16..23}; lanes 16-31 row r,
// k in {8..15,24..31}.
__device__ __forceinline__ v16bf load_tile16(const bf16* src, int ld) {
  const int lane = threadIdx.x & 31;
  const int r    = lane & 15;
  const int kh   = (lane >> 4) << 3;
  const bf16* p  = src + r * ld + kh;
  v16bf v;
#pragma unroll
  for (int i = 0; i < 8; ++i) { v[i] = p[i]; v[8 + i] = p[16 + i]; }
  return v;
}

// Same fragment pattern but loading f32 and converting to bf16 (for the
// loop-invariant w_hh fragments hoisted into VGPRs).
__device__ __forceinline__ v16bf load_tile16_f32cvt(const float* src, int ld) {
  const int lane = threadIdx.x & 31;
  const int r    = lane & 15;
  const int kh   = (lane >> 4) << 3;
  const float* p = src + r * ld + kh;
  v16bf v;
#pragma unroll
  for (int i = 0; i < 8; ++i) { v[i] = (bf16)p[i]; v[8 + i] = (bf16)p[16 + i]; }
  return v;
}

__device__ __forceinline__ v8f wmma_bf16(v16bf a, v16bf b, v8f c) {
  return __builtin_amdgcn_wmma_f32_16x16x32_bf16(false, a, false, b, (short)0, c,
                                                 false, false);
}

// Sense-reversing grid barrier (16 resident blocks).
__device__ __forceinline__ void grid_barrier(volatile unsigned* cnt,
                                             volatile unsigned* gen,
                                             unsigned nb) {
  __syncthreads();
  if (threadIdx.x == 0) {
    __threadfence();
    unsigned g = *gen;
    if (atomicAdd((unsigned*)cnt, 1u) == nb - 1u) {
      *cnt = 0u;
      __threadfence();
      atomicAdd((unsigned*)gen, 1u);
    } else {
      while (*gen == g) { __builtin_amdgcn_s_sleep(8); }
    }
  }
  __syncthreads();
}

// ---------------- kernels ----------------

__global__ void init_ws_kernel(unsigned* bar) {
  if (threadIdx.x < 2) bar[threadIdx.x] = 0u;
}

__global__ void cvt_f32_bf16(const float* __restrict__ in, bf16* __restrict__ out,
                             int n) {
  int i = blockIdx.x * blockDim.x + threadIdx.x;
  if (i < n) out[i] = (bf16)in[i];
}

// gx[m, n] = sum_d x[m,d] * w_ih[n,d] + b_ih[n] + b_hh[n]
// M = B*T = 65536, N = 1024, K = 256. One 16x16 tile per wave, 8 waves/block.
__global__ void xproj_gemm(const bf16* __restrict__ xb,   // [B*T, D]
                           const bf16* __restrict__ wih,  // [G, D] (N,K) row-major
                           const float* __restrict__ b_ih,
                           const float* __restrict__ b_hh,
                           bf16* __restrict__ gx) {       // [B*T, G] bf16
  const int ntiles = Gsz / 16;  // 64
  int wv   = threadIdx.x >> 5;
  int tile = blockIdx.x * 8 + wv;
  int mt   = tile / ntiles;
  int nt   = tile - mt * ntiles;
  int row0 = mt * 16, n0 = nt * 16;

  v8f acc = {};
#pragma unroll
  for (int kk = 0; kk < Dsz / 32; ++kk) {
    v16bf a = load_tile16(xb  + (size_t)row0 * Dsz + kk * 32, Dsz);
    v16bf b = load_tile16(wih + (size_t)n0   * Dsz + kk * 32, Dsz);
    acc = wmma_bf16(a, b, acc);
  }

  int lane  = threadIdx.x & 31;
  int col   = n0 + (lane & 15);
  float bias = b_ih[col] + b_hh[col];
  int rbase = row0 + ((lane >> 4) << 3);
#pragma unroll
  for (int r = 0; r < 8; ++r)
    gx[(size_t)(rbase + r) * Gsz + col] = (bf16)(acc[r] + bias);
}

// Persistent BiLSTM recurrence. 16 blocks: dir = blk>>3, w = blk&7 owns hidden
// units [w*32, w*32+32). The w_hh slice is hoisted into VGPRs as bf16 WMMA
// fragments (loop-invariant across all 2048 steps); h ping-pongs through
// global (L2-resident, 16 KB); c stays in LDS.
__global__ void bilstm_rec(const float* __restrict__ whh_f,
                           const float* __restrict__ whh_b,
                           const bf16*  __restrict__ gx_f,
                           const bf16*  __restrict__ gx_b,
                           const int*   __restrict__ lengths,
                           bf16*        __restrict__ hbuf,   // [2dir][2ping][32][256]
                           float*       __restrict__ feats,  // [B*T, 512]
                           unsigned*    __restrict__ bar) {
  __shared__ float gbuf[32][128];     // gate pre-activations for owned slice
  __shared__ float cbuf[32][32];      // cell state for owned hidden units
  __shared__ int   len_s[32];

  const int tid = threadIdx.x;
  const int dir = blockIdx.x >> 3;
  const int w   = blockIdx.x & 7;
  const float* whh = dir ? whh_b : whh_f;
  const bf16*  gx  = dir ? gx_b  : gx_f;
  bf16* hb = hbuf + (size_t)dir * 2 * Bsz * Hsz;

  const int wv   = tid >> 5;          // 8 waves; wave wv owns local N-tile wv
  const int lane = tid & 31;

  // -- hoist the wave's 16 w_hh columns as 8 K-fragments in registers --
  // local n = gate*32 + hu; tile wv covers global rows
  //   gate*256 + w*32 + (wv&1)*16 ... +16   (contiguous!)
  v16bf bfr[8];
  {
    int gate = wv >> 1;
    int hu0  = (wv & 1) * 16;
    const float* bsrc = whh + (size_t)(gate * Hsz + w * 32 + hu0) * Hsz;
#pragma unroll
    for (int kk = 0; kk < Hsz / 32; ++kk)
      bfr[kk] = load_tile16_f32cvt(bsrc + kk * 32, Hsz);
  }

  // -- preamble: zero cell state and h_{-1} (ping buffer 0), stage lengths --
  for (int e = tid; e < Bsz * 32; e += 256) cbuf[e >> 5][e & 31] = 0.0f;
  for (int e = tid; e < Bsz * 32; e += 256) {
    int bb = e >> 5, hu = e & 31;
    hb[bb * Hsz + w * 32 + hu] = (bf16)0.0f;
  }
  if (tid < 32) len_s[tid] = lengths[tid];
  __syncthreads();
  grid_barrier(bar, bar + 1, 16u);

  for (int t = 0; t < Tsz; ++t) {
    const bf16* hprev = hb + (t & 1) * (Bsz * Hsz);
    bf16*       hnext = hb + ((t + 1) & 1) * (Bsz * Hsz);

    if (t + 1 < Tsz)  // speculative prefetch of next step's gx row segment
      __builtin_prefetch(gx + ((size_t)(tid >> 3) * Tsz + t + 1) * Gsz + (tid & 7) * 128,
                         0, 1);

    v8f acc0 = {}, acc1 = {};         // M-tiles 0 (rows 0-15) and 1 (rows 16-31)
#pragma unroll
    for (int kk = 0; kk < Hsz / 32; ++kk) {
      v16bf a0 = load_tile16(hprev + kk * 32, Hsz);
      v16bf a1 = load_tile16(hprev + 16 * Hsz + kk * 32, Hsz);
      acc0 = wmma_bf16(a0, bfr[kk], acc0);
      acc1 = wmma_bf16(a1, bfr[kk], acc1);
    }

    // store tiles into gbuf, fusing the gx (x-projection + biases) add
    {
      int nloc = wv * 16 + (lane & 15);          // [0,128)
      int gate = nloc >> 5, hu = nloc & 31;
      int gcol = gate * Hsz + w * 32 + hu;       // column in 1024-wide gate space
      int rb   = (lane >> 4) << 3;
#pragma unroll
      for (int r = 0; r < 8; ++r) {
        int bb = rb + r;
        int tg = (dir == 0) ? t : ((t < len_s[bb]) ? (len_s[bb] - 1 - t) : t);
        float g0 = (float)gx[((size_t)bb * Tsz + tg) * Gsz + gcol];
        gbuf[bb][nloc] = acc0[r] + g0;
        int bb1 = bb + 16;
        int tg1 = (dir == 0) ? t : ((t < len_s[bb1]) ? (len_s[bb1] - 1 - t) : t);
        float g1 = (float)gx[((size_t)bb1 * Tsz + tg1) * Gsz + gcol];
        gbuf[bb1][nloc] = acc1[r] + g1;
      }
    }
    __syncthreads();

    // elementwise LSTM cell update for owned 32 hidden units x 32 batch rows
#pragma unroll
    for (int e = 0; e < 4; ++e) {
      int idx = e * 256 + tid;                   // [0,1024)
      int bb = idx >> 5, hu = idx & 31;
      float iv = sigf(gbuf[bb][hu]);
      float fv = sigf(gbuf[bb][32 + hu]);
      float gv = tanhf(gbuf[bb][64 + hu]);
      float ov = sigf(gbuf[bb][96 + hu]);
      float c  = fv * cbuf[bb][hu] + iv * gv;
      cbuf[bb][hu] = c;
      float h = ov * tanhf(c);
      hnext[bb * Hsz + w * 32 + hu] = (bf16)h;
      int tout = (dir == 0) ? t : ((t < len_s[bb]) ? (len_s[bb] - 1 - t) : t);
      feats[((size_t)bb * Tsz + tout) * HOUTsz + dir * Hsz + w * 32 + hu] = h;
    }
    grid_barrier(bar, bar + 1, 16u);             // publish h_t before t+1 reads
  }
}

// emis[m, k] = feats[m, :] . w_tag[k, :] + b_tag[k]   (bandwidth bound, N=8).
// The 64 KB feats tile is staged into LDS with gfx1250 async global->LDS
// loads (GLOBAL_LOAD_ASYNC_TO_LDS_B128, tracked by ASYNCcnt).
__global__ void emis_kernel(const float* __restrict__ feats,
                            const float* __restrict__ w_tag,
                            const float* __restrict__ b_tag,
                            float* __restrict__ emis) {
  __shared__ float ftile[32][512];
  size_t m0 = (size_t)blockIdx.x * 32;
  {
    // LDS byte offset = low 32 bits of the flat shared-aperture address
    unsigned lds_base = (unsigned)(uintptr_t)(void*)&ftile[0][0];
    const float* gsrc = feats + m0 * HOUTsz;
    for (int i = threadIdx.x; i < (32 * 512) / 4; i += 256) {  // 16B chunks
      unsigned loff = lds_base + (unsigned)i * 16u;
      unsigned long long gaddr = (unsigned long long)(uintptr_t)(gsrc + (size_t)i * 4);
      asm volatile("global_load_async_to_lds_b128 %0, %1, off"
                   :: "v"(loff), "v"(gaddr)
                   : "memory");
    }
    asm volatile("s_wait_asynccnt 0x0" ::: "memory");
  }
  __syncthreads();
  int m = threadIdx.x >> 3, k = threadIdx.x & 7;
  float acc = b_tag[k];
  for (int d = 0; d < HOUTsz; ++d) acc += ftile[m][d] * w_tag[k * HOUTsz + d];
  emis[(m0 + m) * Ksz + k] = acc;
}

// CRF numerator + forward algorithm (logZ) + final mean. One block, thread
// (b, j) with b = tid>>3, j = tid&7.
__global__ void crf_kernel(const float* __restrict__ emis,
                           const int*   __restrict__ tags,
                           const int*   __restrict__ lengths,
                           const float* __restrict__ start_trans,
                           const float* __restrict__ end_trans,
                           const float* __restrict__ trans,
                           float* __restrict__ out) {
  __shared__ float score[32][8];
  __shared__ float tr[8][8];
  __shared__ float et[8];
  __shared__ float numpart[32][8];
  __shared__ float res[32];
  __shared__ int   len_s[32];

  int tid = threadIdx.x;
  int b = tid >> 3, j = tid & 7;
  if (tid < 64) (&tr[0][0])[tid] = trans[tid];
  if (tid < 8)  et[tid] = end_trans[tid];
  if (tid < 32) len_s[tid] = lengths[tid];
  __syncthreads();

  // numerator partial sums (t >= 1, masked by length), strided over j
  float np = 0.0f;
  for (int t = 1 + j; t < Tsz; t += 8) {
    if (t < len_s[b]) {
      int tp = tags[b * Tsz + t - 1];
      int tc = tags[b * Tsz + t];
      np += tr[tp][tc] + emis[((size_t)b * Tsz + t) * Ksz + tc];
    }
  }
  numpart[b][j] = np;
  score[b][j] = start_trans[j] + emis[((size_t)b * Tsz) * Ksz + j];
  __syncthreads();

  // sequential forward scan
  for (int t = 1; t < Tsz; ++t) {
    float nxt = 0.0f;
    bool act = t < len_s[b];
    if (act) {
      float e = emis[((size_t)b * Tsz + t) * Ksz + j];
      float mx = -1e30f;
#pragma unroll
      for (int i = 0; i < 8; ++i) mx = fmaxf(mx, score[b][i] + tr[i][j]);
      float s = 0.0f;
#pragma unroll
      for (int i = 0; i < 8; ++i) s += __expf(score[b][i] + tr[i][j] - mx);
      nxt = mx + __logf(s) + e;
    }
    __syncthreads();
    if (act) score[b][j] = nxt;
    __syncthreads();
  }

  if (j == 0) {
    float mx = -1e30f;
    for (int i = 0; i < 8; ++i) mx = fmaxf(mx, score[b][i] + et[i]);
    float s = 0.0f;
    for (int i = 0; i < 8; ++i) s += __expf(score[b][i] + et[i] - mx);
    float logZ = mx + __logf(s);
    float num = 0.0f;
    for (int i = 0; i < 8; ++i) num += numpart[b][i];
    int t0tag = tags[b * Tsz];
    num += start_trans[t0tag] + emis[((size_t)b * Tsz) * Ksz + t0tag];
    num += et[tags[b * Tsz + len_s[b] - 1]];
    res[b] = num - logZ;
  }
  __syncthreads();
  if (tid == 0) {
    float s = 0.0f;
    for (int bb = 0; bb < 32; ++bb) s += res[bb];
    out[0] = -s / (float)Bsz;
  }
}

// ---------------- launch ----------------

extern "C" void kernel_launch(void* const* d_in, const int* in_sizes, int n_in,
                              void* d_out, int out_size, void* d_ws, size_t ws_size,
                              hipStream_t stream) {
  const float* sentences   = (const float*)d_in[0];
  const int*   tags        = (const int*)  d_in[1];
  const int*   lengths     = (const int*)  d_in[2];
  /* d_in[3] = mask: recomputed from lengths, unused */
  const float* w_ih_f      = (const float*)d_in[4];
  const float* w_hh_f      = (const float*)d_in[5];
  const float* b_ih_f      = (const float*)d_in[6];
  const float* b_hh_f      = (const float*)d_in[7];
  const float* w_ih_b      = (const float*)d_in[8];
  const float* w_hh_b      = (const float*)d_in[9];
  const float* b_ih_b      = (const float*)d_in[10];
  const float* b_hh_b      = (const float*)d_in[11];
  const float* w_tag       = (const float*)d_in[12];
  const float* b_tag       = (const float*)d_in[13];
  const float* start_trans = (const float*)d_in[14];
  const float* end_trans   = (const float*)d_in[15];
  const float* trans       = (const float*)d_in[16];
  float* out = (float*)d_out;

  char* ws = (char*)d_ws;
  size_t off = 0;
  auto walloc = [&](size_t bytes) -> void* {
    void* p = ws + off;
    off += (bytes + 255) & ~(size_t)255;
    return p;
  };
  bf16*     xb    = (bf16*)walloc((size_t)Bsz * Tsz * Dsz * sizeof(bf16));
  bf16*     wfb   = (bf16*)walloc((size_t)Gsz * Dsz * sizeof(bf16));
  bf16*     wbb   = (bf16*)walloc((size_t)Gsz * Dsz * sizeof(bf16));
  bf16*     gx_f  = (bf16*)walloc((size_t)Bsz * Tsz * Gsz * sizeof(bf16));
  bf16*     gx_b  = (bf16*)walloc((size_t)Bsz * Tsz * Gsz * sizeof(bf16));
  float*    feats = (float*)walloc((size_t)Bsz * Tsz * HOUTsz * sizeof(float));
  float*    emis  = (float*)walloc((size_t)Bsz * Tsz * Ksz * sizeof(float));
  bf16*     hbuf  = (bf16*)walloc((size_t)2 * 2 * Bsz * Hsz * sizeof(bf16));
  unsigned* bar   = (unsigned*)walloc(256);

  init_ws_kernel<<<1, 64, 0, stream>>>(bar);

  int nx = Bsz * Tsz * Dsz;          // 16.8M
  cvt_f32_bf16<<<(nx + 255) / 256, 256, 0, stream>>>(sentences, xb, nx);
  int nw = Gsz * Dsz;                // 262144
  cvt_f32_bf16<<<(nw + 255) / 256, 256, 0, stream>>>(w_ih_f, wfb, nw);
  cvt_f32_bf16<<<(nw + 255) / 256, 256, 0, stream>>>(w_ih_b, wbb, nw);

  // 65536/16 M-tiles * 64 N-tiles / 8 waves = 32768 blocks per direction
  xproj_gemm<<<32768, 256, 0, stream>>>(xb, wfb, b_ih_f, b_hh_f, gx_f);
  xproj_gemm<<<32768, 256, 0, stream>>>(xb, wbb, b_ih_b, b_hh_b, gx_b);

  bilstm_rec<<<16, 256, 0, stream>>>(w_hh_f, w_hh_b, gx_f, gx_b, lengths, hbuf,
                                     feats, bar);

  emis_kernel<<<(Bsz * Tsz) / 32, 256, 0, stream>>>(feats, w_tag, b_tag, emis);

  crf_kernel<<<1, 256, 0, stream>>>(emis, tags, lengths, start_trans, end_trans,
                                    trans, out);
}
